// FeatureAttention_19533511262570
// MI455X (gfx1250) — matrix operand order repaired and checked by reading, not compiled, not run on hardware
//
#include <hip/hip_runtime.h>

typedef __attribute__((ext_vector_type(16))) _Float16 v16h;
typedef __attribute__((ext_vector_type(8)))  _Float16 v8h;
typedef __attribute__((ext_vector_type(8)))  float    v8f;

#define CC   256          // channels
#define CH   64           // bottleneck channels (C/R)
#define BB   1024         // segments
#define RPB  256          // rows per block in segment reduction

// ---------------- init segment buffers ----------------
__global__ void fa_init(float* __restrict__ sumbuf, float* __restrict__ maxbuf) {
    int i = blockIdx.x * 256 + threadIdx.x;       // 0 .. B*C-1
    sumbuf[i] = 0.0f;
    maxbuf[i] = -__builtin_inff();
}

// float atomic max via monotonic int transform
__device__ __forceinline__ void atomicMaxF(float* addr, float v) {
    if (v >= 0.0f) atomicMax((int*)addr, __float_as_int(v));
    else           atomicMin((unsigned int*)addr, (unsigned int)__float_as_int(v));
}

// ---------------- segment sum + max (batch is sorted), float4 per thread -----
__global__ void fa_segreduce(const float* __restrict__ x, const int* __restrict__ batch,
                             float* __restrict__ sumbuf, float* __restrict__ maxbuf, int N) {
    const int t = threadIdx.x;                    // 0..63 : float4 chunk (channels 4t..4t+3)
    long long r0 = (long long)blockIdx.x * RPB;
    long long r1 = r0 + RPB; if (r1 > N) r1 = N;
    if (r0 >= r1) return;

    const float4* x4 = (const float4*)x;
    const float ninf = -__builtin_inff();
    float4 s = {0.f, 0.f, 0.f, 0.f};
    float4 m = {ninf, ninf, ninf, ninf};
    int cur = batch[r0];
    for (long long r = r0; r < r1; ++r) {
        int seg = batch[r];
        if (seg != cur) {
            long long base = (long long)cur * CC + t * 4;
            atomicAdd(&sumbuf[base + 0], s.x); atomicAdd(&sumbuf[base + 1], s.y);
            atomicAdd(&sumbuf[base + 2], s.z); atomicAdd(&sumbuf[base + 3], s.w);
            atomicMaxF(&maxbuf[base + 0], m.x); atomicMaxF(&maxbuf[base + 1], m.y);
            atomicMaxF(&maxbuf[base + 2], m.z); atomicMaxF(&maxbuf[base + 3], m.w);
            s = (float4){0.f, 0.f, 0.f, 0.f};
            m = (float4){ninf, ninf, ninf, ninf};
            cur = seg;
        }
        float4 v = x4[r * (CC / 4) + t];          // global_load_b128, coalesced
        s.x += v.x; s.y += v.y; s.z += v.z; s.w += v.w;
        m.x = fmaxf(m.x, v.x); m.y = fmaxf(m.y, v.y);
        m.z = fmaxf(m.z, v.z); m.w = fmaxf(m.w, v.w);
    }
    long long base = (long long)cur * CC + t * 4;
    atomicAdd(&sumbuf[base + 0], s.x); atomicAdd(&sumbuf[base + 1], s.y);
    atomicAdd(&sumbuf[base + 2], s.z); atomicAdd(&sumbuf[base + 3], s.w);
    atomicMaxF(&maxbuf[base + 0], m.x); atomicMaxF(&maxbuf[base + 1], m.y);
    atomicMaxF(&maxbuf[base + 2], m.z); atomicMaxF(&maxbuf[base + 3], m.w);
}

// ---------------- replace -inf sentinel (empty segments) with 0 --------------
__global__ void fa_fixmax(float* __restrict__ maxbuf) {
    int i = blockIdx.x * 256 + threadIdx.x;       // 0 .. B*C-1
    float v = maxbuf[i];
    maxbuf[i] = (v == -__builtin_inff()) ? 0.0f : v;
}

// ---------------- MLP stage 1: hidden = relu(concat(max,sum) @ w1) -> f16 ----
// one wave per 16x16 tile; K = 256 -> 8 WMMAs, fully unrolled
__global__ void fa_mlp1(const float* __restrict__ maxbuf, const float* __restrict__ sumbuf,
                        const float* __restrict__ w1, _Float16* __restrict__ hidden) {
    const int mt   = blockIdx.x;                  // 0..127  (2048/16 row tiles)
    const int nt   = blockIdx.y;                  // 0..3
    const int lane = threadIdx.x;

    const int am     = lane & 15;
    const int akq    = (lane < 16) ? 0 : 2;       // akbase/4 : float4 index
    const int bn     = lane & 15;
    const int bkbase = (lane < 16) ? 0 : 8;

    // block-uniform select: first 64 row-tiles read maxbuf, rest read sumbuf
    const float* rowp = (mt < BB / 16)
        ? (maxbuf + (size_t)(mt * 16 + am) * CC)
        : (sumbuf + (size_t)((mt - BB / 16) * 16 + am) * CC);

    v8f acc = {};
#pragma unroll
    for (int kt = 0; kt < CC / 32; ++kt) {
        const float4* p = (const float4*)(rowp + kt * 32);
        float4 lo0 = p[akq],     lo1 = p[akq + 1];      // k = kt*32 + akbase + 0..7
        float4 hi0 = p[akq + 4], hi1 = p[akq + 5];      // k = kt*32 + 16 + akbase + 0..7
        v16h a;
        a[0]  = (_Float16)lo0.x; a[1]  = (_Float16)lo0.y;
        a[2]  = (_Float16)lo0.z; a[3]  = (_Float16)lo0.w;
        a[4]  = (_Float16)lo1.x; a[5]  = (_Float16)lo1.y;
        a[6]  = (_Float16)lo1.z; a[7]  = (_Float16)lo1.w;
        a[8]  = (_Float16)hi0.x; a[9]  = (_Float16)hi0.y;
        a[10] = (_Float16)hi0.z; a[11] = (_Float16)hi0.w;
        a[12] = (_Float16)hi1.x; a[13] = (_Float16)hi1.y;
        a[14] = (_Float16)hi1.z; a[15] = (_Float16)hi1.w;
        v16h b;
#pragma unroll
        for (int h = 0; h < 16; ++h)
            b[h] = (_Float16)w1[(size_t)(kt * 32 + bkbase + (h & 7) + ((h >> 3) << 4)) * CH + nt * 16 + bn];
        acc = __builtin_amdgcn_wmma_f32_16x16x32_f16(false, a, false, b,
                                                     (short)0, acc, false, false);
    }
    const int n = lane & 15, mb = (lane < 16) ? 0 : 8;
#pragma unroll
    for (int r = 0; r < 8; ++r) {
        float v = acc[r];
        hidden[(size_t)(mt * 16 + mb + r) * CH + nt * 16 + n] =
            (_Float16)(v > 0.0f ? v : 0.0f);
    }
}

// ---------------- MLP stage 2: outbuf = hidden @ w2 (no relu) ----------------
__global__ void fa_mlp2(const _Float16* __restrict__ hidden, const float* __restrict__ w2,
                        float* __restrict__ outbuf) {
    const int mt   = blockIdx.x;                  // 0..127
    const int nt   = blockIdx.y;                  // 0..15
    const int lane = threadIdx.x;

    const int am     = lane & 15;
    const int akbase = (lane < 16) ? 0 : 8;
    const int bn     = lane & 15;
    const int bkbase = (lane < 16) ? 0 : 8;

    const _Float16* rowp = hidden + (size_t)(mt * 16 + am) * CH;

    v8f acc = {};
#pragma unroll
    for (int kt = 0; kt < CH / 32; ++kt) {        // K = 64 -> 2 WMMAs
        v8h lo = *(const v8h*)(rowp + kt * 32 + akbase);       // 16B vector load
        v8h hi = *(const v8h*)(rowp + kt * 32 + 16 + akbase);
        v16h a;
#pragma unroll
        for (int h = 0; h < 8; ++h) { a[h] = lo[h]; a[h + 8] = hi[h]; }
        v16h b;
#pragma unroll
        for (int h = 0; h < 16; ++h)
            b[h] = (_Float16)w2[(size_t)(kt * 32 + bkbase + (h & 7) + ((h >> 3) << 4)) * CC + nt * 16 + bn];
        acc = __builtin_amdgcn_wmma_f32_16x16x32_f16(false, a, false, b,
                                                     (short)0, acc, false, false);
    }
    const int n = lane & 15, mb = (lane < 16) ? 0 : 8;
#pragma unroll
    for (int r = 0; r < 8; ++r)
        outbuf[(size_t)(mt * 16 + mb + r) * CC + nt * 16 + n] = acc[r];
}

// ---------------- gather + gate: out = x * relu(o_max[b] + o_sum[b]) ---------
__global__ void fa_gate(const float* __restrict__ x, const int* __restrict__ batch,
                        const float* __restrict__ outbuf, float* __restrict__ out,
                        long long total4) {
    long long tid = (long long)blockIdx.x * 256 + threadIdx.x;  // one float4 each
    if (tid >= total4) return;
    long long row = tid >> 6;                     // 64 float4 per row
    int c4 = (int)(tid & 63);
    int b = batch[row];

    float4 xv = ((const float4*)x)[row * 64 + c4];
    float4 tv = ((const float4*)(outbuf + (long long)b * CC))[c4];          // mlp(max)
    float4 uv = ((const float4*)(outbuf + (long long)(b + BB) * CC))[c4];   // mlp(sum)
    float4 rv;
    rv.x = xv.x * fmaxf(tv.x + uv.x, 0.0f);
    rv.y = xv.y * fmaxf(tv.y + uv.y, 0.0f);
    rv.z = xv.z * fmaxf(tv.z + uv.z, 0.0f);
    rv.w = xv.w * fmaxf(tv.w + uv.w, 0.0f);
    ((float4*)out)[row * 64 + c4] = rv;
}

extern "C" void kernel_launch(void* const* d_in, const int* in_sizes, int n_in,
                              void* d_out, int out_size, void* d_ws, size_t ws_size,
                              hipStream_t stream) {
    const float* x     = (const float*)d_in[0];
    const int*   batch = (const int*)d_in[1];
    const float* w1    = (const float*)d_in[2];
    const float* w2    = (const float*)d_in[3];
    float*       out   = (float*)d_out;
    const int N = in_sizes[1];                    // number of nodes

    // workspace layout (bytes):
    //   sumbuf  [B*C]   f32 : 1 MB
    //   maxbuf  [B*C]   f32 : 1 MB
    //   hidden  [2B*CH] f16 : 256 KB
    //   outbuf  [2B*C]  f32 : 2 MB
    char* ws = (char*)d_ws;
    float*    sumbuf = (float*)ws;
    float*    maxbuf = (float*)(ws + (size_t)BB * CC * 4);
    _Float16* hidden = (_Float16*)(ws + (size_t)2 * BB * CC * 4);
    float*    outbuf = (float*)(ws + (size_t)2 * BB * CC * 4 + (size_t)2 * BB * CH * 2);

    fa_init<<<(BB * CC) / 256, 256, 0, stream>>>(sumbuf, maxbuf);

    int nblk = (N + RPB - 1) / RPB;
    fa_segreduce<<<nblk, 64, 0, stream>>>(x, batch, sumbuf, maxbuf, N);

    fa_fixmax<<<(BB * CC) / 256, 256, 0, stream>>>(maxbuf);

    fa_mlp1<<<dim3(2 * BB / 16, CH / 16), 32, 0, stream>>>(maxbuf, sumbuf, w1, hidden);
    fa_mlp2<<<dim3(2 * BB / 16, CC / 16), 32, 0, stream>>>(hidden, w2, outbuf);

    long long total4 = (long long)N * (CC / 4);
    fa_gate<<<(unsigned)((total4 + 255) / 256), 256, 0, stream>>>(x, batch, outbuf, out, total4);
}
